// Transformer_7155415515783
// MI455X (gfx1250) — compile-verified, matrix operand order checked by reading, not run on hardware
//
#include <hip/hip_runtime.h>

// ---------------------------------------------------------------------------
// DynamicViT forward for MI455X (gfx1250, wave32, WMMA bf16, TDM-fed LDS)
// B=128, N=197, C=384, H=6, HD=64, D=12, prune at 3/6/9
// All GEMM operands are bf16 in memory; TDM (tensor_load_to_lds) streams
// 128x32 tiles into LDS (double buffered) while waves run WMMA.
// ---------------------------------------------------------------------------

typedef __attribute__((ext_vector_type(16))) __bf16        v16bf;
typedef __attribute__((ext_vector_type(8)))  float         v8f;
typedef __attribute__((ext_vector_type(4)))  unsigned int  u32x4;
typedef __attribute__((ext_vector_type(8)))  int           i32x8;
typedef __attribute__((ext_vector_type(4)))  int           i32x4;

union BFrag { uint4 u[2]; v16bf v; };

#define LDSW 40         // halves per LDS tile row: 32 data + 8 pad = 80B pitch
#define TILEB 10240     // one 128x32 bf16 tile with padding, bytes

__device__ inline unsigned short f2bf(float f) {
  unsigned int u = __float_as_uint(f);
  u += 0x7FFFu + ((u >> 16) & 1u);          // round-to-nearest-even
  return (unsigned short)(u >> 16);
}
__device__ inline float wsum(float v) {
#pragma unroll
  for (int o = 16; o > 0; o >>= 1) v += __shfl_xor(v, o, 32);
  return v;
}
__device__ inline float wmaxf(float v) {
#pragma unroll
  for (int o = 16; o > 0; o >>= 1) v = fmaxf(v, __shfl_xor(v, o, 32));
  return v;
}
__device__ inline float gelu_exact(float x) {
  return 0.5f * x * (1.0f + erff(x * 0.7071067811865475f));
}

// ---------------------------------------------------------------------------
// TDM: issue an async 2D bf16 tile load (tile_d0 x tile_d1, row-major) from
// global into LDS at byte offset lds_off. LDS rows get 16B padding every 64B
// so the staged tile has an 80B pitch (LDSW). OOB rows/cols are zero-filled
// by the TDM (tensor_dim0/1), which handles all M/N/K tails for free.
// Issued by one wave; completion tracked with TENSORcnt.
// ---------------------------------------------------------------------------
__device__ inline void tdm_load_tile(unsigned lds_off, const void* gaddr,
                                     unsigned tensor_d0, unsigned tensor_d1,
                                     unsigned tile_d0, unsigned tile_d1,
                                     unsigned long long stride_elems)
{
  const unsigned long long ga = (unsigned long long)(size_t)gaddr;
  u32x4 g0;
  g0[0] = 1u;                                        // count=1 (valid user D#)
  g0[1] = lds_off;                                   // lds_addr (bytes)
  g0[2] = (unsigned)(ga & 0xFFFFFFFFu);              // global_addr[31:0]
  g0[3] = (unsigned)((ga >> 32) & 0x01FFFFFFu)       // global_addr[56:32]
        | 0x80000000u;                               // type=2 ("image")
  i32x8 g1;
  g1[0] = (int)((1u << 16)                           // data_size = 2 bytes
              | (1u << 20)                           // pad_enable
              | (3u << 22)                           // pad_interval: 16 DWORDs
              | (3u << 25));                         // pad_amount:    4 DWORDs
  g1[1] = (int)((tensor_d0 & 0xFFFFu) << 16);                         // [63:48]
  g1[2] = (int)(((tensor_d0 >> 16) & 0xFFFFu) | ((tensor_d1 & 0xFFFFu) << 16));
  g1[3] = (int)(((tensor_d1 >> 16) & 0xFFFFu) | ((tile_d0 & 0xFFFFu) << 16));
  g1[4] = (int)(tile_d1 & 0xFFFFu);                  // tile_dim1, tile_dim2=0
  g1[5] = (int)(unsigned)(stride_elems & 0xFFFFFFFFull);   // dim0 stride lo
  g1[6] = (int)(unsigned)((stride_elems >> 32) & 0xFFFFull);
  g1[7] = 0;
  const i32x4 z4 = {0, 0, 0, 0};
#if defined(__clang_major__) && (__clang_major__ >= 23)
  const i32x8 z8 = {0, 0, 0, 0, 0, 0, 0, 0};
  __builtin_amdgcn_tensor_load_to_lds(g0, g1, z4, z4, z8, 0);
#else
  __builtin_amdgcn_tensor_load_to_lds(g0, g1, z4, z4, 0);
#endif
}

// ---------------------------------------------------------------------------
// Generic batched bf16 GEMM:  C[z] = scale * A[z](MxK) * W[z]^T(NxK) + bias
// A and W are bf16; K % 32 == 0. Both tiles are fetched by the TDM with
// double buffering; the waves only do ds_load fragments + WMMA.
// MODE: 0 = bias, 1 = bias+GELU, 2 = bias+residual(f32)
// OUTBF: output bf16 (else f32)
// dynamic LDS: 4 tiles = 40960 bytes
// ---------------------------------------------------------------------------
template <int MODE, bool OUTBF>
__global__ void __launch_bounds__(256)
gemm_nt_kernel(const unsigned short* __restrict__ Ag,
               const unsigned short* __restrict__ Wg,
               const float* __restrict__ bias, const float* __restrict__ Rg,
               void* __restrict__ Cg,
               int M, int N, int K, int lda, int ldw, int ldc, float scale,
               int Hdiv, long long bA, long long hA, long long bW, long long hW,
               long long bC, long long hC)
{
  extern __shared__ char smem[];   // [buf0: A|B][buf1: A|B]

  const int z  = blockIdx.z;
  const int zb = z / Hdiv;
  const int zh = z - zb * Hdiv;
  const unsigned short* A = Ag + (size_t)zb * bA + (size_t)zh * hA;
  const unsigned short* W = Wg + (size_t)zb * bW + (size_t)zh * hW;
  float*          Cf = (float*)Cg + (size_t)zb * bC + (size_t)zh * hC;
  unsigned short* Cb = (unsigned short*)Cg + (size_t)zb * bC + (size_t)zh * hC;
  const float* R = (MODE == 2) ? (Rg + (size_t)zb * bC + (size_t)zh * hC) : nullptr;

  const int m0 = blockIdx.y * 128;
  const int n0 = blockIdx.x * 128;
  const int tid  = threadIdx.x;
  const int wave = tid >> 5;
  const int lane = tid & 31;
  const int wm = wave & 1;             // 2 wave rows of 64
  const int wn = wave >> 1;            // 4 wave cols of 32
  const int lm   = lane & 15;
  const int koff = (lane >> 4) << 3;   // 0 | 8 halves (wave32 WMMA layout)

  v8f acc[4][2];
#pragma unroll
  for (int i = 0; i < 4; i++)
#pragma unroll
    for (int j = 0; j < 2; j++)
#pragma unroll
      for (int e = 0; e < 8; e++) acc[i][j][e] = 0.0f;

  const int nIter = K >> 5;
  if (wave == 0) {                      // prologue: first A/B tiles
    tdm_load_tile(0,     A + (size_t)m0 * lda, (unsigned)K, (unsigned)(M - m0),
                  32, 128, (unsigned long long)lda);
    tdm_load_tile(TILEB, W + (size_t)n0 * ldw, (unsigned)K, (unsigned)(N - n0),
                  32, 128, (unsigned long long)ldw);
  }

  for (int it = 0; it < nIter; it++) {
    if (wave == 0) __builtin_amdgcn_s_wait_tensorcnt(0);
    __syncthreads();                    // tiles for `it` visible to all waves
    if (wave == 0 && it + 1 < nIter) {  // stream next tiles into other buffer
      const int k1 = (it + 1) << 5;
      const unsigned boff = ((it + 1) & 1) * (2 * TILEB);
      tdm_load_tile(boff,         A + (size_t)m0 * lda + k1,
                    (unsigned)(K - k1), (unsigned)(M - m0), 32, 128,
                    (unsigned long long)lda);
      tdm_load_tile(boff + TILEB, W + (size_t)n0 * ldw + k1,
                    (unsigned)(K - k1), (unsigned)(N - n0), 32, 128,
                    (unsigned long long)ldw);
    }
    const unsigned short* As = (const unsigned short*)(smem + (it & 1) * (2 * TILEB));
    const unsigned short* Bs = (const unsigned short*)(smem + (it & 1) * (2 * TILEB) + TILEB);

    BFrag a[4], bfr[2];
#pragma unroll
    for (int mi = 0; mi < 4; mi++) {
      const int r = (wm * 64 + mi * 16 + lm) * LDSW;
      a[mi].u[0] = *(const uint4*)&As[r + koff];
      a[mi].u[1] = *(const uint4*)&As[r + 16 + koff];
    }
#pragma unroll
    for (int ni = 0; ni < 2; ni++) {
      const int r = (wn * 32 + ni * 16 + lm) * LDSW;
      bfr[ni].u[0] = *(const uint4*)&Bs[r + koff];
      bfr[ni].u[1] = *(const uint4*)&Bs[r + 16 + koff];
    }
#pragma unroll
    for (int mi = 0; mi < 4; mi++)
#pragma unroll
      for (int ni = 0; ni < 2; ni++)
        acc[mi][ni] = __builtin_amdgcn_wmma_f32_16x16x32_bf16(
            false, a[mi].v, false, bfr[ni].v, (short)0, acc[mi][ni], false, false);
  }

  // epilogue: lanes 0-15 -> M=e, lanes 16-31 -> M=8+e, N=lane&15
  const int rl = (lane >> 4) << 3;
#pragma unroll
  for (int ni = 0; ni < 2; ni++) {
    const int col = n0 + wn * 32 + ni * 16 + lm;
    if (col >= N) continue;
    const float bv = bias ? bias[col] : 0.0f;
#pragma unroll
    for (int mi = 0; mi < 4; mi++) {
#pragma unroll
      for (int e = 0; e < 8; e++) {
        const int row = m0 + wm * 64 + mi * 16 + rl + e;
        if (row < M) {
          float v = acc[mi][ni][e] * scale + bv;
          if (MODE == 1) v = gelu_exact(v);
          if (MODE == 2) v += R[(size_t)row * ldc + col];
          if (OUTBF) Cb[(size_t)row * ldc + col] = f2bf(v);
          else       Cf[(size_t)row * ldc + col] = v;
        }
      }
    }
  }
}

// ---------------------------------------------------------------------------
// Attention A*V:  o[b,h] (197x64) = probs[b,h] (197x224, bf16, zero-padded)
//                                  @ v[b,h] (197x64, bf16)
// A tile via TDM (double buffered); V transpose-staged by the waves with
// clamped unconditional loads (no branches). K = 224 (7 steps of 32).
// dynamic LDS: 2*A tile + 2*B tile = 30720 bytes
// ---------------------------------------------------------------------------
__global__ void __launch_bounds__(256)
attn_av_kernel(const unsigned short* __restrict__ probs,
               const unsigned short* __restrict__ qkv,
               unsigned short* __restrict__ out)
{
  extern __shared__ char smem[];   // As0 | As1 | Bs0 | Bs1

  const int z  = blockIdx.z;            // b*6 + h
  const int zb = z / 6;
  const int zh = z - zb * 6;
  const unsigned short* Ap = probs + (size_t)z * (197 * 224);
  const unsigned short* V  = qkv + (size_t)zb * (197 * 1152) + 768 + zh * 64;
  const int m0 = blockIdx.y * 128;

  const int tid  = threadIdx.x;
  const int wave = tid >> 5;
  const int lane = tid & 31;
  const int wm = wave >> 1;             // 0..3 -> 32-row groups
  const int wn = wave & 1;              // 0..1 -> 32-col groups
  const int lm   = lane & 15;
  const int koff = (lane >> 4) << 3;

  v8f acc[2][2];
#pragma unroll
  for (int i = 0; i < 2; i++)
#pragma unroll
    for (int j = 0; j < 2; j++)
#pragma unroll
      for (int e = 0; e < 8; e++) acc[i][j][e] = 0.0f;

  if (wave == 0)
    tdm_load_tile(0, Ap + (size_t)m0 * 224, 224u, (unsigned)(197 - m0),
                  32, 128, 224ull);

  for (int it = 0; it < 7; it++) {
    const int k0 = it << 5;
    // ---- waves transpose-stage V tile: Bs[n][k] (clamped, branch-free) ----
    {
      unsigned short* Bs = (unsigned short*)(smem + 2 * TILEB + (it & 1) * 5120);
      const int kr = tid >> 3;                 // 0..31
      const int nb = (tid & 7) << 3;           // 0,8,..,56
      const int kk = k0 + kr;
      const int kc = (kk < 196) ? kk : 196;    // clamp: load is always in-bounds
      uint4 u = *(const uint4*)(V + (size_t)kc * 1152 + nb);
      if (kk >= 197) u = make_uint4(0u, 0u, 0u, 0u);   // zero pad (selects)
#pragma unroll
      for (int i = 0; i < 4; i++) {
        const unsigned w = (&u.x)[i];
        Bs[(nb + 2 * i)     * LDSW + kr] = (unsigned short)(w & 0xFFFFu);
        Bs[(nb + 2 * i + 1) * LDSW + kr] = (unsigned short)(w >> 16);
      }
    }
    if (wave == 0) __builtin_amdgcn_s_wait_tensorcnt(0);
    __syncthreads();
    if (wave == 0 && it < 6)
      tdm_load_tile(((it + 1) & 1) * TILEB, Ap + (size_t)m0 * 224 + k0 + 32,
                    (unsigned)(224 - k0 - 32), (unsigned)(197 - m0),
                    32, 128, 224ull);

    const unsigned short* As = (const unsigned short*)(smem + (it & 1) * TILEB);
    const unsigned short* Bs = (const unsigned short*)(smem + 2 * TILEB + (it & 1) * 5120);

    BFrag a[2], bfr[2];
#pragma unroll
    for (int mi = 0; mi < 2; mi++) {
      const int r = (wm * 32 + mi * 16 + lm) * LDSW;
      a[mi].u[0] = *(const uint4*)&As[r + koff];
      a[mi].u[1] = *(const uint4*)&As[r + 16 + koff];
    }
#pragma unroll
    for (int ni = 0; ni < 2; ni++) {
      const int r = (wn * 32 + ni * 16 + lm) * LDSW;
      bfr[ni].u[0] = *(const uint4*)&Bs[r + koff];
      bfr[ni].u[1] = *(const uint4*)&Bs[r + 16 + koff];
    }
#pragma unroll
    for (int mi = 0; mi < 2; mi++)
#pragma unroll
      for (int ni = 0; ni < 2; ni++)
        acc[mi][ni] = __builtin_amdgcn_wmma_f32_16x16x32_bf16(
            false, a[mi].v, false, bfr[ni].v, (short)0, acc[mi][ni], false, false);
    __syncthreads();   // Bs[it&1] fully consumed before restaged at it+2
  }

  const int rl = (lane >> 4) << 3;
#pragma unroll
  for (int ni = 0; ni < 2; ni++) {
    const int col = wn * 32 + ni * 16 + lm;            // < 64
#pragma unroll
    for (int mi = 0; mi < 2; mi++) {
#pragma unroll
      for (int e = 0; e < 8; e++) {
        const int row = m0 + wm * 32 + mi * 16 + rl + e;
        if (row < 197)
          out[((size_t)zb * 197 + row) * 384 + zh * 64 + col] = f2bf(acc[mi][ni][e]);
      }
    }
  }
}

// ---------------------------------------------------------------------------
// LayerNorm: one wave32 per row, bf16 output (feeds GEMM A via TDM)
// ---------------------------------------------------------------------------
__global__ void __launch_bounds__(256)
ln_kernel(const float* __restrict__ in, unsigned short* __restrict__ out,
          const float* __restrict__ w, const float* __restrict__ b,
          int rows, int C, float eps)
{
  const int wave = threadIdx.x >> 5, lane = threadIdx.x & 31;
  const int row = blockIdx.x * 8 + wave;
  if (row >= rows) return;
  const float* rp = in + (size_t)row * C;
  float v[12];
  const int nit = (C + 31) >> 5;
  float s = 0.0f;
  for (int t = 0; t < nit; t++) {
    const int c = lane + (t << 5);
    v[t] = (c < C) ? rp[c] : 0.0f;
    s += v[t];
  }
  const float mean = wsum(s) / (float)C;
  float vs = 0.0f;
  for (int t = 0; t < nit; t++) {
    const int c = lane + (t << 5);
    const float d = (c < C) ? (v[t] - mean) : 0.0f;
    vs += d * d;
  }
  const float rstd = rsqrtf(wsum(vs) / (float)C + eps);
  unsigned short* op = out + (size_t)row * C;
  for (int t = 0; t < nit; t++) {
    const int c = lane + (t << 5);
    if (c < C) op[c] = f2bf((v[t] - mean) * rstd * w[c] + b[c]);
  }
}

// ---------------------------------------------------------------------------
// Policy softmax (eps=0, diagonal forced to 1): one wave32 per row.
// Reads f32 logits (pitch 197), writes bf16 probs zero-padded to pitch 224.
// ---------------------------------------------------------------------------
__global__ void __launch_bounds__(256)
softmax_policy_kernel(const float* __restrict__ attn, const float* __restrict__ pol,
                      unsigned short* __restrict__ probs)
{
  const int wave = threadIdx.x >> 5, lane = threadIdx.x & 31;
  const int row = blockIdx.x * 8 + wave;
  if (row >= 128 * 6 * 197) return;
  const int n  = row % 197;
  const int bh = row / 197;
  const int b  = bh / 6;
  const float* rp = attn + (size_t)bh * (197 * 197) + (size_t)n * 197;
  const float* pp = pol + (size_t)b * 197;
  unsigned short* op = probs + (size_t)bh * (197 * 224) + (size_t)n * 224;
  float x[7], ap[7];
  float mx = -1e30f;
#pragma unroll
  for (int t = 0; t < 7; t++) {
    const int m = lane + (t << 5);
    if (m < 197) { x[t] = rp[m]; ap[t] = (m == n) ? 1.0f : pp[m]; mx = fmaxf(mx, x[t]); }
    else         { x[t] = -1e30f; ap[t] = 0.0f; }
  }
  mx = wmaxf(mx);
  float s = 0.0f;
#pragma unroll
  for (int t = 0; t < 7; t++) {
    const int m = lane + (t << 5);
    if (m < 197) { x[t] = __expf(x[t] - mx) * ap[t]; s += x[t]; }
  }
  const float inv = 1.0f / wsum(s);
#pragma unroll
  for (int t = 0; t < 7; t++) {
    const int m = lane + (t << 5);
    op[m] = (m < 197) ? f2bf(x[t] * inv) : (unsigned short)0;   // pad 197..223
  }
}

// ---------------------------------------------------------------------------
// Predictor stage 1: per (b,n,head): LN(64, eps 1e-5) -> 64x64 matmul -> GELU
// ---------------------------------------------------------------------------
__global__ void __launch_bounds__(256)
pred_stage1_kernel(const float* __restrict__ tok,
                   const float* __restrict__ lnw, const float* __restrict__ lnb,
                   const float* __restrict__ inw, const float* __restrict__ inb,
                   float* __restrict__ hout)
{
  __shared__ float sv[8][64];
  const int wave = threadIdx.x >> 5, lane = threadIdx.x & 31;
  const int rid = blockIdx.x * 8 + wave;
  const int total = 128 * 196 * 6;
  const bool act = rid < total;
  const float* xr = tok;
  if (act) {
    const int head = rid % 6;
    const int t = rid / 6;
    const int n  = t % 196;
    const int bb = t / 196;
    xr = tok + ((size_t)bb * 197 + 1 + n) * 384 + head * 64;
  }
  const float x0 = act ? xr[lane] : 0.0f;
  const float x1 = act ? xr[lane + 32] : 0.0f;
  const float mean = wsum(x0 + x1) * (1.0f / 64.0f);
  const float d0 = x0 - mean, d1 = x1 - mean;
  const float var = wsum(d0 * d0 + d1 * d1) * (1.0f / 64.0f);
  const float rs = rsqrtf(var + 1e-5f);
  sv[wave][lane]      = d0 * rs * lnw[lane]      + lnb[lane];
  sv[wave][lane + 32] = d1 * rs * lnw[lane + 32] + lnb[lane + 32];
  __syncthreads();
  if (act) {
#pragma unroll
    for (int half = 0; half < 2; half++) {
      const int d = lane + (half << 5);
      float a = inb[d];
      const float* wr = inw + (size_t)d * 64;
      for (int k = 0; k < 64; k++) a += sv[wave][k] * wr[k];
      hout[(size_t)rid * 64 + d] = gelu_exact(a);
    }
  }
}

// Predictor global pooling: glob[b,head,d2] = sum_n h[...,32+d2]*prev / sum prev
__global__ void __launch_bounds__(256)
pred_reduce_kernel(const float* __restrict__ h, const float* __restrict__ prev,
                   float* __restrict__ glob)
{
  const int wave = threadIdx.x >> 5, lane = threadIdx.x & 31;
  const int rid = blockIdx.x * 8 + wave;
  if (rid >= 128 * 6 * 32) return;
  const int d2 = rid % 32;
  const int t  = rid / 32;
  const int head = t % 6;
  const int bb   = t / 6;
  float num = 0.0f, den = 0.0f;
  for (int n = lane; n < 196; n += 32) {
    const float p = prev[(size_t)bb * 196 + n];
    num += h[(((size_t)bb * 196 + n) * 6 + head) * 64 + 32 + d2] * p;
    den += p;
  }
  num = wsum(num);
  den = wsum(den);
  if (lane == 0) glob[((size_t)bb * 6 + head) * 32 + d2] = num / den;
}

// Predictor head MLP: [local|glob] -> 32 -> 16 -> 2 -> logsoftmax, mean heads
__global__ void __launch_bounds__(256)
pred_stage2_kernel(const float* __restrict__ h, const float* __restrict__ glob,
                   const float* __restrict__ o1w, const float* __restrict__ o1b,
                   const float* __restrict__ o2w, const float* __restrict__ o2b,
                   const float* __restrict__ o3w, const float* __restrict__ o3b,
                   float* __restrict__ score)
{
  __shared__ float vec[8][64];
  __shared__ float y1[8][32];
  __shared__ float y2[8][16];
  const int wave = threadIdx.x >> 5, lane = threadIdx.x & 31;
  const int rid = blockIdx.x * 8 + wave;
  const int total = 128 * 196;
  const int n  = rid % 196;
  const int bb = rid / 196;
  float s0acc = 0.0f, s1acc = 0.0f;
  for (int head = 0; head < 6; head++) {
    if (rid < total) {
      vec[wave][lane]      = h[(((size_t)bb * 196 + n) * 6 + head) * 64 + lane];
      vec[wave][lane + 32] = glob[((size_t)bb * 6 + head) * 32 + lane];
    }
    __syncthreads();
    {
      float a = o1b[lane];
      const float* wr = o1w + (size_t)lane * 64;
      for (int k = 0; k < 64; k++) a += vec[wave][k] * wr[k];
      y1[wave][lane] = gelu_exact(a);
    }
    __syncthreads();
    if (lane < 16) {
      float a = o2b[lane];
      const float* wr = o2w + (size_t)lane * 32;
      for (int k = 0; k < 32; k++) a += y1[wave][k] * wr[k];
      y2[wave][lane] = gelu_exact(a);
    }
    __syncthreads();
    float s0 = o3b[0], s1 = o3b[1];
    for (int k = 0; k < 16; k++) {
      const float yv = y2[wave][k];
      s0 += yv * o3w[k];
      s1 += yv * o3w[16 + k];
    }
    const float m = fmaxf(s0, s1);
    const float lse = m + logf(__expf(s0 - m) + __expf(s1 - m));
    s0acc += (s0 - lse) * (1.0f / 6.0f);
    s1acc += (s1 - lse) * (1.0f / 6.0f);
    __syncthreads();
  }
  if (rid < total && lane == 0) {
    score[(size_t)rid * 2]     = s0acc;
    score[(size_t)rid * 2 + 1] = s1acc;
  }
}

// Gumbel straight-through decision: y == hard one-hot in forward value
__global__ void __launch_bounds__(256)
pred_decide_kernel(const float* __restrict__ score, const float* __restrict__ gum,
                   float* __restrict__ prev, float* __restrict__ pol)
{
  const int idx = blockIdx.x * blockDim.x + threadIdx.x;
  if (idx >= 128 * 196) return;
  const int n  = idx % 196;
  const int bb = idx / 196;
  const float u0 = gum[(size_t)idx * 2];
  const float u1 = gum[(size_t)idx * 2 + 1];
  const float g0 = -logf(-logf(u0 + 1e-10f) + 1e-10f);
  const float g1 = -logf(-logf(u1 + 1e-10f) + 1e-10f);
  const float s0 = score[(size_t)idx * 2] + g0;
  const float s1 = score[(size_t)idx * 2 + 1] + g1;
  const float keep = (s0 >= s1) ? prev[idx] : 0.0f;   // argmax ties -> index 0
  prev[idx] = keep;
  pol[(size_t)bb * 197 + 1 + n] = keep;
  if (n == 0) pol[(size_t)bb * 197] = 1.0f;
}

// ---------------------------------------------------------------------------
// Layout glue
// ---------------------------------------------------------------------------
__global__ void __launch_bounds__(256)
cvt_bf16_kernel(const float* __restrict__ src, unsigned short* __restrict__ dst, int n)
{
  const int idx = blockIdx.x * blockDim.x + threadIdx.x;
  if (idx < n) dst[idx] = f2bf(src[idx]);
}

__global__ void __launch_bounds__(256)
assemble_tokens_kernel(const float* __restrict__ x, const float* __restrict__ cls,
                       float* __restrict__ tok)
{
  const int idx = blockIdx.x * blockDim.x + threadIdx.x;
  if (idx >= 128 * 197 * 384) return;
  const int c  = idx % 384;
  const int t  = (idx / 384) % 197;
  const int bb = idx / (384 * 197);
  tok[idx] = (t == 0) ? cls[(size_t)bb * 384 + c]
                      : x[((size_t)bb * 384 + c) * 196 + (t - 1)];
}

__global__ void __launch_bounds__(256)
init_policy_kernel(const float* __restrict__ pin, float* __restrict__ pol,
                   float* __restrict__ prev)
{
  const int idx = blockIdx.x * blockDim.x + threadIdx.x;
  if (idx >= 128 * 197) return;
  const float v = pin[idx];
  pol[idx] = v;
  const int t = idx % 197;
  const int bb = idx / 197;
  if (t > 0) prev[(size_t)bb * 196 + (t - 1)] = v;
}

__global__ void __launch_bounds__(256)
write_out_kernel(const float* __restrict__ tok, float* __restrict__ out)
{
  const int idx = blockIdx.x * blockDim.x + threadIdx.x;
  if (idx >= 128 * 197 * 384) return;
  const int c  = idx % 384;
  const int t  = (idx / 384) % 197;
  const int bb = idx / (384 * 197);
  const float v = tok[idx];
  if (t == 0) out[(size_t)(128 * 384 * 196) + (size_t)bb * 384 + c] = v;  // cls
  else        out[((size_t)bb * 384 + c) * 196 + (t - 1)] = v;            // sp
}

// ---------------------------------------------------------------------------
extern "C" void kernel_launch(void* const* d_in, const int* in_sizes, int n_in,
                              void* d_out, int out_size, void* d_ws, size_t ws_size,
                              hipStream_t stream)
{
  (void)in_sizes; (void)n_in; (void)out_size; (void)ws_size;
  const float* x    = (const float*)d_in[0];
  const float* cls  = (const float*)d_in[1];
  const float* pin  = (const float*)d_in[2];
  const float* gum  = (const float*)d_in[3];
  const float* ln1w = (const float*)d_in[4];
  const float* ln1b = (const float*)d_in[5];
  const float* qkvw = (const float*)d_in[6];
  const float* qkvb = (const float*)d_in[7];
  const float* projw= (const float*)d_in[8];
  const float* projb= (const float*)d_in[9];
  const float* ln2w = (const float*)d_in[10];
  const float* ln2b = (const float*)d_in[11];
  const float* fc1w = (const float*)d_in[12];
  const float* fc1b = (const float*)d_in[13];
  const float* fc2w = (const float*)d_in[14];
  const float* fc2b = (const float*)d_in[15];
  const float* plnw = (const float*)d_in[16];
  const float* plnb = (const float*)d_in[17];
  const float* pinw = (const float*)d_in[18];
  const float* pinb = (const float*)d_in[19];
  const float* po1w = (const float*)d_in[20];
  const float* po1b = (const float*)d_in[21];
  const float* po2w = (const float*)d_in[22];
  const float* po2b = (const float*)d_in[23];
  const float* po3w = (const float*)d_in[24];
  const float* po3b = (const float*)d_in[25];

  char* ws = (char*)d_ws;
  size_t off = 0;
  auto carveB = [&](size_t bytes) -> void* {
    void* p = ws + off;
    off += ((bytes + 255) & ~(size_t)255);
    return p;
  };
  const size_t TOKN = (size_t)25216 * 384;                  // B*N*C
  float*  tokA  = (float*)carveB(TOKN * 4);
  float*  tokB  = (float*)carveB(TOKN * 4);
  unsigned short* hbuf = (unsigned short*)carveB(TOKN * 2); // bf16 LN out
  unsigned short* obuf = (unsigned short*)carveB(TOKN * 2); // bf16 attn out
  unsigned short* wqkv = (unsigned short*)carveB((size_t)12*1152*384 * 2);
  unsigned short* wproj= (unsigned short*)carveB((size_t)12*384*384 * 2);
  unsigned short* wfc1 = (unsigned short*)carveB((size_t)12*1536*384 * 2);
  unsigned short* wfc2 = (unsigned short*)carveB((size_t)12*384*1536 * 2);
  float*  glob  = (float*)carveB((size_t)128*6*32 * 4);
  float*  score = (float*)carveB((size_t)128*196*2 * 4);
  float*  pol   = (float*)carveB((size_t)128*197 * 4);
  float*  prev  = (float*)carveB((size_t)128*196 * 4);
  const size_t bigoff = off;
  unsigned short* qkvbuf = (unsigned short*)carveB((size_t)25216*1152 * 2);
  float*          attnbuf= (float*)carveB((size_t)768*197*197 * 4);
  unsigned short* probs  = (unsigned short*)carveB((size_t)768*197*224 * 2);
  // disjoint lifetimes -> alias the big region (qkv/attn dead during MLP):
  unsigned short* mlp1  = (unsigned short*)(ws + bigoff);   // B*N*1536 bf16
  float*          predh = (float*)(ws + bigoff);            // B*196*6*64 f32

  const dim3 blk(256);

  // one-time (per launch) weight f32 -> bf16 conversion
  cvt_bf16_kernel<<<(12*1152*384 + 255)/256, blk, 0, stream>>>(qkvw, wqkv, 12*1152*384);
  cvt_bf16_kernel<<<(12*384*384  + 255)/256, blk, 0, stream>>>(projw, wproj, 12*384*384);
  cvt_bf16_kernel<<<(12*1536*384 + 255)/256, blk, 0, stream>>>(fc1w, wfc1, 12*1536*384);
  cvt_bf16_kernel<<<(12*384*1536 + 255)/256, blk, 0, stream>>>(fc2w, wfc2, 12*384*1536);

  assemble_tokens_kernel<<<(128*197*384 + 255)/256, blk, 0, stream>>>(x, cls, tokA);
  init_policy_kernel<<<(128*197 + 255)/256, blk, 0, stream>>>(pin, pol, prev);

  int pc = 0;
  for (int i = 0; i < 12; i++) {
    if (i == 3 || i == 6 || i == 9) {
      pred_stage1_kernel<<<128*196*6/8, blk, 0, stream>>>(
          tokA, plnw + pc*64, plnb + pc*64, pinw + (size_t)pc*64*64, pinb + pc*64, predh);
      pred_reduce_kernel<<<128*6*32/8, blk, 0, stream>>>(predh, prev, glob);
      pred_stage2_kernel<<<128*196/8, blk, 0, stream>>>(
          predh, glob, po1w + (size_t)pc*32*64, po1b + pc*32,
          po2w + (size_t)pc*16*32, po2b + pc*16,
          po3w + (size_t)pc*2*16, po3b + pc*2, score);
      pred_decide_kernel<<<(128*196 + 255)/256, blk, 0, stream>>>(
          score, gum + (size_t)pc*128*196*2, prev, pol);
      pc++;
    }
    // ---- attention half ----
    ln_kernel<<<25216/8, blk, 0, stream>>>(tokA, hbuf, ln1w + i*384, ln1b + i*384,
                                           25216, 384, 1e-6f);
    gemm_nt_kernel<0, true><<<dim3(9, 197, 1), blk, 40960, stream>>>(
        hbuf, wqkv + (size_t)i*1152*384, qkvb + i*1152, nullptr, qkvbuf,
        25216, 1152, 384, 384, 384, 1152, 1.0f, 1, 0, 0, 0, 0, 0, 0);
    gemm_nt_kernel<0, false><<<dim3(2, 2, 768), blk, 40960, stream>>>( // q k^T/8
        qkvbuf, qkvbuf + 384, nullptr, nullptr, attnbuf,
        197, 197, 64, 1152, 1152, 197, 0.125f,
        6, 197LL*1152, 64LL, 197LL*1152, 64LL, 6LL*197*197, 197LL*197);
    softmax_policy_kernel<<<128*6*197/8, blk, 0, stream>>>(attnbuf, pol, probs);
    attn_av_kernel<<<dim3(1, 2, 768), blk, 30720, stream>>>(probs, qkvbuf, obuf);
    gemm_nt_kernel<2, false><<<dim3(3, 197, 1), blk, 40960, stream>>>( // +res
        obuf, wproj + (size_t)i*384*384, projb + i*384, tokA, tokB,
        25216, 384, 384, 384, 384, 384, 1.0f, 1, 0, 0, 0, 0, 0, 0);
    // ---- MLP half ----
    ln_kernel<<<25216/8, blk, 0, stream>>>(tokB, hbuf, ln2w + i*384, ln2b + i*384,
                                           25216, 384, 1e-6f);
    gemm_nt_kernel<1, true><<<dim3(12, 197, 1), blk, 40960, stream>>>( // GELU
        hbuf, wfc1 + (size_t)i*1536*384, fc1b + i*1536, nullptr, mlp1,
        25216, 1536, 384, 384, 384, 1536, 1.0f, 1, 0, 0, 0, 0, 0, 0);
    gemm_nt_kernel<2, false><<<dim3(3, 197, 1), blk, 40960, stream>>>( // +res
        mlp1, wfc2 + (size_t)i*384*1536, fc2b + i*384, tokB, tokA,
        25216, 384, 1536, 1536, 1536, 384, 1.0f, 1, 0, 0, 0, 0, 0, 0);
  }

  write_out_kernel<<<(128*197*384 + 255)/256, blk, 0, stream>>>(tokA, (float*)d_out);
}
